// MiniTriplane_69140383531570
// MI455X (gfx1250) — matrix operand order repaired and checked by reading, not compiled, not run on hardware
//
#include <hip/hip_runtime.h>

#define PTS_PER_BLOCK 128
#define KPAD 224      // 198 features padded to 7*32 for K=32 WMMA steps
#define HIDN 128
#define INDIM 198

typedef _Float16 v16h __attribute__((ext_vector_type(16)));
typedef _Float16 v8h  __attribute__((ext_vector_type(8)));
typedef float    v8f  __attribute__((ext_vector_type(8)));
typedef float    f2v  __attribute__((ext_vector_type(2)));

union V16 { v16h v; v8h h[2]; };

// Load two adjacent floats with only 4-byte alignment guaranteed.
__device__ __forceinline__ f2v ldpair(const float* p) {
  f2v v;
  __builtin_memcpy(&v, p, sizeof(f2v));
  return v;
}

// Bilinear sample channels [c0,c1) of a (32,128,128) plane. For g in [-1,1],
// clamping the cell index to [0, dim-2] is exactly equivalent to the
// reference's zero-weighted OOB taps.
__device__ __forceinline__ void plane_feat(const float* __restrict__ pl,
                                           float gx, float gy,
                                           int c0, int c1,
                                           _Float16* __restrict__ dst) {
  const int H = 128, W = 128, HW = 16384;
  float ix = (gx + 1.f) * 0.5f * (float)(W - 1);
  float iy = (gy + 1.f) * 0.5f * (float)(H - 1);
  int x0 = (int)floorf(ix); x0 = min(max(x0, 0), W - 2);
  int y0 = (int)floorf(iy); y0 = min(max(y0, 0), H - 2);
  float wx1 = ix - (float)x0, wx0 = 1.f - wx1;
  float wy1 = iy - (float)y0, wy0 = 1.f - wy1;
  float w00 = wx0 * wy0, w01 = wx1 * wy0, w10 = wx0 * wy1, w11 = wx1 * wy1;
  int base0 = y0 * W + x0;
  int base1 = base0 + W;
  for (int cg = c0; cg < c1; cg += 8) {
    v8h outv;
#pragma unroll
    for (int c = 0; c < 8; ++c) {
      const float* p = pl + (cg + c) * HW;
      f2v a = ldpair(p + base0);
      f2v b = ldpair(p + base1);
      float acc = fmaf(w00, a.x, fmaf(w01, a.y, fmaf(w10, b.x, w11 * b.y)));
      outv[c] = (_Float16)acc;
    }
    *(v8h*)(dst + (cg - c0)) = outv;
  }
}

// 1D line sample: reference calls bilinear on (32,128,1) with gx==gy; the
// W-dimension taps collapse (W=1), leaving a 1D lerp over H=128.
__device__ __forceinline__ void line_feat(const float* __restrict__ fl,
                                          float g, _Float16* __restrict__ dst) {
  const int H = 128;
  float iy = (g + 1.f) * 0.5f * (float)(H - 1);
  int y0 = (int)floorf(iy); y0 = min(max(y0, 0), H - 2);
  float wy1 = iy - (float)y0, wy0 = 1.f - wy1;
  for (int cg = 0; cg < 32; cg += 8) {
    v8h outv;
#pragma unroll
    for (int c = 0; c < 8; ++c) {
      f2v v = ldpair(fl + (cg + c) * H + y0);
      outv[c] = (_Float16)fmaf(wy0, v.x, wy1 * v.y);
    }
    *(v8h*)(dst + cg) = outv;
  }
}

__device__ __forceinline__ void vol_feat(const float* __restrict__ vol,
                                         float gx, float gy, float gz,
                                         _Float16* __restrict__ dst) {
  const int D = 64, H = 64, W = 64;
  float ix = (gx + 1.f) * 0.5f * (float)(W - 1);
  float iy = (gy + 1.f) * 0.5f * (float)(H - 1);
  float iz = (gz + 1.f) * 0.5f * (float)(D - 1);
  int x0 = (int)floorf(ix); x0 = min(max(x0, 0), W - 2);
  int y0 = (int)floorf(iy); y0 = min(max(y0, 0), H - 2);
  int z0 = (int)floorf(iz); z0 = min(max(z0, 0), D - 2);
  float wx1 = ix - (float)x0, wx0 = 1.f - wx1;
  float wy1 = iy - (float)y0, wy0 = 1.f - wy1;
  float wz1 = iz - (float)z0, wz0 = 1.f - wz1;
  int b00 = (z0 * H + y0) * W + x0;       // (z0 ,y0 )
  int b10 = b00 + W;                      // (z0 ,y0+1)
  int b01 = b00 + H * W;                  // (z0+1,y0 )
  int b11 = b01 + W;                      // (z0+1,y0+1)
  float w00 = wy0 * wz0, w10 = wy1 * wz0, w01 = wy0 * wz1, w11 = wy1 * wz1;
#pragma unroll
  for (int c = 0; c < 6; ++c) {
    const float* p = vol + c * (D * H * W);
    f2v a = ldpair(p + b00);
    f2v b = ldpair(p + b10);
    f2v e = ldpair(p + b01);
    f2v f = ldpair(p + b11);
    float lo = fmaf(w00, fmaf(wx0, a.x, wx1 * a.y),
                    w10 * fmaf(wx0, b.x, wx1 * b.y));
    float hi = fmaf(w01, fmaf(wx0, e.x, wx1 * e.y),
                    w11 * fmaf(wx0, f.x, wx1 * f.y));
    dst[c] = (_Float16)(lo + hi);
  }
}

__global__ void __launch_bounds__(256)
triplane_mlp_kernel(const float* __restrict__ coords,
                    const float* __restrict__ planes,
                    const float* __restrict__ lines,
                    const float* __restrict__ volume,
                    const float* __restrict__ w1,
                    const float* __restrict__ b1,
                    const float* __restrict__ w2,
                    const float* __restrict__ b2,
                    float* __restrict__ out, int npts) {
  extern __shared__ char smem[];
  _Float16* feats = (_Float16*)smem;                 // [128][224] fp16
  _Float16* w1t   = feats + PTS_PER_BLOCK * KPAD;    // [128][224] fp16 (w1^T)
  float*    b1s   = (float*)(w1t + HIDN * KPAD);     // [128]
  float*    w2s   = b1s + HIDN;                      // [128]

  const int tid = threadIdx.x;

  // ---- Stage w1 (transposed, fp16, K zero-padded), b1, w2 into LDS ----
  for (int i = tid; i < KPAD * HIDN; i += 256) {
    int k = i >> 7;          // 0..223
    int n = i & (HIDN - 1);  // 0..127
    float v = (k < INDIM) ? w1[k * HIDN + n] : 0.f;
    w1t[n * KPAD + k] = (_Float16)v;
  }
  if (tid < HIDN)            b1s[tid] = b1[tid];
  else if (tid < 2 * HIDN)   w2s[tid - HIDN] = w2[tid - HIDN];

  // ---- Phase 1: features for a 128-point tile, 2 threads/point ----
  {
    int pl = tid >> 1;        // local point 0..127
    int half = tid & 1;
    int p = blockIdx.x * PTS_PER_BLOCK + pl;
    _Float16* fr = feats + pl * KPAD;
    if (p < npts) {
      float x = coords[p * 3 + 0];
      float y = coords[p * 3 + 1];
      float z = coords[p * 3 + 2];
      if (half == 0) {
        // lines (x_e,y_e,z_e), volume, xz-plane ch 0..15, K padding
        line_feat(lines + 0 * 32 * 128, x, fr + 0);
        line_feat(lines + 1 * 32 * 128, y, fr + 32);
        line_feat(lines + 2 * 32 * 128, z, fr + 64);
        vol_feat(volume, x, y, z, fr + 192);
        plane_feat(planes + 2 * 32 * 16384, x, z, 0, 16, fr + 160);
#pragma unroll
        for (int k = INDIM; k < KPAD; ++k) fr[k] = (_Float16)0.f;
      } else {
        // xy-plane, yz-plane, xz-plane ch 16..31
        plane_feat(planes + 0 * 32 * 16384, x, y, 0, 32, fr + 96);
        plane_feat(planes + 1 * 32 * 16384, y, z, 0, 32, fr + 128);
        plane_feat(planes + 2 * 32 * 16384, x, z, 16, 32, fr + 176);
      }
    } else {
      if (half == 0) {
        for (int k = 0; k < 96; ++k)     fr[k] = (_Float16)0.f;
        for (int k = 160; k < 176; ++k)  fr[k] = (_Float16)0.f;
        for (int k = 192; k < KPAD; ++k) fr[k] = (_Float16)0.f;
      } else {
        for (int k = 96; k < 160; ++k)   fr[k] = (_Float16)0.f;
        for (int k = 176; k < 192; ++k)  fr[k] = (_Float16)0.f;
      }
    }
  }
  __syncthreads();

  // ---- Phase 2: fused MLP. Wave wv owns points [16*wv, 16*wv+16). ----
  const int wv   = tid >> 5;
  const int lane = tid & 31;
  const int r    = lane & 15;   // A row / B col within tile
  const int hsel = lane >> 4;   // lane-half selects K sub-pattern

  // Preload all 7 A fragments (16x32 fp16).
  // 16-bit A layout: lane half 0 holds K {0..7,16..23}, half 1 {8..15,24..31}.
  const _Float16* frow = feats + (wv * 16 + r) * KPAD;
  V16 afr[7];
#pragma unroll
  for (int kk = 0; kk < 7; ++kk) {
    afr[kk].h[0] = *(const v8h*)(frow + kk * 32 + hsel * 8);
    afr[kk].h[1] = *(const v8h*)(frow + kk * 32 + 16 + hsel * 8);
  }

  // 8 independent accumulators (one per N-tile) -> no WMMA->WMMA RAW chain;
  // K-outer order lets B loads for later tiles overlap in-flight WMMAs.
  v8f acc[8];
#pragma unroll
  for (int n = 0; n < 8; ++n) acc[n] = (v8f){};

#pragma unroll
  for (int kk = 0; kk < 7; ++kk) {
#pragma unroll
    for (int n = 0; n < 8; ++n) {
      // 16-bit B layout: lane half 0 holds K {0..7,16..23}, half 1 {8..15,24..31} (matches A).
      const _Float16* bbase = w1t + (n * 16 + r) * KPAD + kk * 32 + hsel * 8;
      V16 bfr;
      bfr.h[0] = *(const v8h*)(bbase);
      bfr.h[1] = *(const v8h*)(bbase + 16);
      acc[n] = __builtin_amdgcn_wmma_f32_16x16x32_f16(false, afr[kk].v,
                                                      false, bfr.v,
                                                      (short)0, acc[n],
                                                      false, false);
    }
  }

  // Epilogue: bias + relu + layer-2 dot with w2.
  float partial[8];
#pragma unroll
  for (int v = 0; v < 8; ++v) partial[v] = 0.f;
#pragma unroll
  for (int n = 0; n < 8; ++n) {
    int col = n * 16 + r;
    float bb = b1s[col];
    float ww = w2s[col];
#pragma unroll
    for (int v = 0; v < 8; ++v) {
      float h = acc[n][v] + bb;
      h = fmaxf(h, 0.f);
      partial[v] = fmaf(h, ww, partial[v]);
    }
  }

  // Reduce across the 16 lanes of each half (halves hold different rows).
#pragma unroll
  for (int off = 1; off < 16; off <<= 1)
#pragma unroll
    for (int v = 0; v < 8; ++v)
      partial[v] += __shfl_xor(partial[v], off, 32);

  if (r == 0) {  // lane 0 -> rows 0..7, lane 16 -> rows 8..15
    float bias2 = b2[0];
    int pb = blockIdx.x * PTS_PER_BLOCK + wv * 16 + hsel * 8;
#pragma unroll
    for (int v = 0; v < 8; ++v) {
      int p = pb + v;
      if (p < npts) out[p] = partial[v] + bias2;
    }
  }
}

extern "C" void kernel_launch(void* const* d_in, const int* in_sizes, int n_in,
                              void* d_out, int out_size, void* d_ws, size_t ws_size,
                              hipStream_t stream) {
  const float* coords = (const float*)d_in[0];
  const float* planes = (const float*)d_in[1];
  const float* lines  = (const float*)d_in[2];
  const float* volume = (const float*)d_in[3];
  const float* w1     = (const float*)d_in[4];
  const float* b1     = (const float*)d_in[5];
  const float* w2     = (const float*)d_in[6];
  const float* b2     = (const float*)d_in[7];
  float* out = (float*)d_out;

  int npts = in_sizes[0] / 3;
  int nblocks = (npts + PTS_PER_BLOCK - 1) / PTS_PER_BLOCK;
  size_t smem = (size_t)(PTS_PER_BLOCK * KPAD + HIDN * KPAD) * sizeof(_Float16)
              + 2 * HIDN * sizeof(float);  // ~113 KB, within 320 KB/WGP

  hipFuncSetAttribute((const void*)triplane_mlp_kernel,
                      hipFuncAttributeMaxDynamicSharedMemorySize, (int)smem);

  triplane_mlp_kernel<<<nblocks, 256, smem, stream>>>(
      coords, planes, lines, volume, w1, b1, w2, b2, out, npts);
}
